// MultiheadAttention_70901320123130
// MI455X (gfx1250) — matrix operand order repaired and checked by reading, not compiled, not run on hardware
//
#include <hip/hip_runtime.h>
#include <hip/hip_bf16.h>
#include <math.h>

// ---------------------------------------------------------------------------
// MI455X / gfx1250 multi-head attention forward (data + probs outputs).
// All GEMMs: v_wmma_f32_16x16x32_bf16 (wave32, 16x16 C tiles, f32 accum).
// Staging: float2 loads + v_perm_b32 packing; raw bf16 copies via
// global_load_async_to_lds_b128 (ASYNCcnt-tracked).
// Problem dims are fixed: B=2, T=S=2048, D=NH=2048, N=16 heads, H=128.
// ---------------------------------------------------------------------------

typedef __attribute__((ext_vector_type(16))) __bf16        v16bf;
typedef __attribute__((ext_vector_type(8)))  float         v8f;
typedef __attribute__((ext_vector_type(8)))  unsigned int  v8u;
typedef __attribute__((ext_vector_type(4)))  int           v4i;

union FragU { v8u u; v16bf bf; };

// f32 -> bf16, round-to-nearest (2 VALU)
static __device__ __forceinline__ unsigned short f2bf(float f) {
  unsigned int x = __builtin_bit_cast(unsigned int, f) + 0x8000u;
  return (unsigned short)(x >> 16);
}

// two f32 -> packed bf16x2: 2 adds + 1 v_perm_b32
static __device__ __forceinline__ unsigned int pk_bf16(float lo, float hi) {
  unsigned int a = __builtin_bit_cast(unsigned int, lo) + 0x8000u;
  unsigned int b = __builtin_bit_cast(unsigned int, hi) + 0x8000u;
#if __has_builtin(__builtin_amdgcn_perm)
  return __builtin_amdgcn_perm(b, a, 0x07060302u);  // {b[3],b[2],a[3],a[2]}
#else
  return (a >> 16) | (b & 0xFFFF0000u);
#endif
}

#if __has_builtin(__builtin_amdgcn_global_load_async_to_lds_b128) && \
    __has_builtin(__builtin_amdgcn_s_wait_asynccnt)
#define HAVE_ASYNC_LDS 1
// 16-byte global -> LDS async copy (ASYNCcnt-tracked, no VGPR round trip).
static __device__ __forceinline__ void async_cp16(unsigned short* dst_lds,
                                                  const unsigned short* src_g) {
  __builtin_amdgcn_global_load_async_to_lds_b128(
      (__attribute__((address_space(1))) v4i*)src_g,
      (__attribute__((address_space(3))) v4i*)dst_lds, 0, 0);
}
static __device__ __forceinline__ void async_wait0() {
  __builtin_amdgcn_s_wait_asynccnt(0);
}
#endif

static __device__ __forceinline__ v8f wmma_bf16(v8u a, v8u b, v8f c) {
  FragU A, B; A.u = a; B.u = b;
  return __builtin_amdgcn_wmma_f32_16x16x32_bf16(
      false, A.bf, false, B.bf, (short)0, c, false, false);
}

// A fragment (16x32 bf16, M x K), LDS row-major M x K (stride ld shorts).
static __device__ __forceinline__ v8u lds_load_a(const unsigned short* base,
                                                 int ld, int m, int k0, int lane) {
  const unsigned short* p = base + (m + (lane & 15)) * ld + k0 + ((lane >> 4) ? 8 : 0);
  v8u u;
  u[0] = *(const unsigned int*)(p + 0);
  u[1] = *(const unsigned int*)(p + 2);
  u[2] = *(const unsigned int*)(p + 4);
  u[3] = *(const unsigned int*)(p + 6);
  u[4] = *(const unsigned int*)(p + 16);
  u[5] = *(const unsigned int*)(p + 18);
  u[6] = *(const unsigned int*)(p + 20);
  u[7] = *(const unsigned int*)(p + 22);
  return u;
}

// B fragment (32x16 bf16, K x N), LDS stored N-major (row-major N x K, stride ld).
static __device__ __forceinline__ v8u lds_load_b_nk(const unsigned short* base,
                                                    int ld, int n, int k0, int lane) {
  const unsigned short* p = base + (n + (lane & 15)) * ld + k0 + ((lane >> 4) ? 8 : 0);
  v8u u;
#pragma unroll
  for (int v = 0; v < 8; ++v) u[v] = *(const unsigned int*)(p + 2 * v + (v >= 4 ? 8 : 0));
  return u;
}

#define LDA 40    // 32 + 8 pad (shorts); 80B rows -> 16B aligned
#define LDB 136   // 128 + 8 pad (shorts); 272B rows -> 16B aligned

// ---------------------------------------------------------------------------
// Kernel 1: QKV projection.  Out = (X * W + bias) * scale, bf16 output.
// X: (4096 x 2048) f32, W: (2048 x 2048) f32 (staged transposed).
// Block = 256 thr (8 waves), tile 128M x 128N, K-step 32.
// ---------------------------------------------------------------------------
__global__ __launch_bounds__(256) void mha_qkv_proj(
    const float* __restrict__ X, const float* __restrict__ W,
    const float* __restrict__ bias, unsigned short* __restrict__ Out,
    float scale) {
  constexpr int K = 2048, N = 2048;
  __shared__ __align__(16) unsigned short As[128 * LDA];   // M x K
  __shared__ __align__(16) unsigned short Bs[128 * LDA];   // N x K (transposed)
  const int t = threadIdx.x, lane = t & 31, w = t >> 5;
  const int wm = w & 3, wn = w >> 2;
  const int m_base = blockIdx.y * 128, n_base = blockIdx.x * 128;
  v8f acc[2][4] = {};

  for (int k0 = 0; k0 < K; k0 += 32) {
    __syncthreads();
#pragma unroll
    for (int i = 0; i < 8; ++i) {  // stage A: 128 x 32, float2 + perm pack
      int e = t + i * 256, r = e >> 4, cp = e & 15;
      float2 v = *(const float2*)(X + (size_t)(m_base + r) * K + k0 + cp * 2);
      *(unsigned int*)(As + r * LDA + cp * 2) = pk_bf16(v.x, v.y);
    }
#pragma unroll
    for (int i = 0; i < 8; ++i) {  // stage B transposed: Bs[n][k]
      int e = t + i * 256, kp = e >> 7, n = e & 127;
      float lo = W[(size_t)(k0 + 2 * kp) * N + n_base + n];
      float hi = W[(size_t)(k0 + 2 * kp + 1) * N + n_base + n];
      *(unsigned int*)(Bs + n * LDA + 2 * kp) = pk_bf16(lo, hi);
    }
    if (k0 + 32 < K) {  // global_prefetch_b8 of next tiles
      __builtin_prefetch(&X[(size_t)(m_base + (t >> 1)) * K + k0 + 32 + (t & 1) * 16], 0, 1);
      __builtin_prefetch(&W[(size_t)(k0 + 32 + (t >> 3)) * N + n_base + (t & 7) * 16], 0, 1);
    }
    __syncthreads();
    v8u a[2], b[4];
#pragma unroll
    for (int mi = 0; mi < 2; ++mi) a[mi] = lds_load_a(As, LDA, wm * 32 + mi * 16, 0, lane);
#pragma unroll
    for (int ni = 0; ni < 4; ++ni) b[ni] = lds_load_b_nk(Bs, LDA, wn * 64 + ni * 16, 0, lane);
#pragma unroll
    for (int mi = 0; mi < 2; ++mi)
#pragma unroll
      for (int ni = 0; ni < 4; ++ni) acc[mi][ni] = wmma_bf16(a[mi], b[ni], acc[mi][ni]);
  }

  const int half8 = (lane >> 4) ? 8 : 0;
#pragma unroll
  for (int ni = 0; ni < 4; ++ni) {
    int col = n_base + wn * 64 + ni * 16 + (lane & 15);
    float bv = bias[col];
    unsigned short* op = Out + (size_t)(m_base + wm * 32 + half8) * N + col;
#pragma unroll
    for (int mi = 0; mi < 2; ++mi)
#pragma unroll
      for (int r = 0; r < 8; ++r)
        op[(size_t)(mi * 16 + r) * N] = f2bf((acc[mi][ni][r] + bv) * scale);
  }
}

// ---------------------------------------------------------------------------
// Kernel 2: logits = Q*K^T + bias; fp32 softmax over S; write probs (f32).
// Q pre-scaled by 1/sqrt(H).  Block handles 32 T-rows of one (b, head).
// Pass 1: streamed online (max, sumexp); Pass 2: recompute + normalized write.
// ---------------------------------------------------------------------------
__global__ __launch_bounds__(256) void mha_attn_softmax(
    const unsigned short* __restrict__ Q, const unsigned short* __restrict__ Kmat,
    const float* __restrict__ bias, float* __restrict__ probs) {
  constexpr int T = 2048, NH = 2048, H = 128;
  __shared__ __align__(16) unsigned short Qs[32 * LDB];
  __shared__ __align__(16) unsigned short Ks[64 * LDB];
  __shared__ float m_red[4][32];
  __shared__ float l_red[4][32];
  const int t = threadIdx.x, lane = t & 31, w = t >> 5;
  const int mg = w >> 2, sg = w & 3;
  const int b = blockIdx.z, n = blockIdx.y;
  const int t0 = blockIdx.x * 32;
  const int half8 = (lane >> 4) ? 8 : 0;
  const unsigned short* Qb = Q + (size_t)b * T * NH + n * H;
  const unsigned short* Kb = Kmat + (size_t)b * T * NH + n * H;

  // per-thread staging addresses (strength-reduced; advance by stride per iter)
  const int sr = t >> 4, sx = (t & 15) * 8;  // staging row / 16B column
  unsigned short* kd = Ks + sr * LDB + sx;

  // stage Q tile (32 x 128 bf16) once
#ifdef HAVE_ASYNC_LDS
  {
    const unsigned short* qp = Qb + (size_t)(t0 + sr) * NH + sx;
#pragma unroll
    for (int i = 0; i < 2; ++i)
      async_cp16(Qs + sr * LDB + sx + i * 16 * LDB, qp + (size_t)i * 16 * NH);
    async_wait0();
  }
#else
#pragma unroll
  for (int i = 0; i < 8; ++i) {
    int e = t + i * 256, r = e >> 6, cu = e & 63;
    *(unsigned int*)(Qs + r * LDB + cu * 2) =
        *(const unsigned int*)(Qb + (size_t)(t0 + r) * NH + cu * 2);
  }
#endif
  __syncthreads();
  v8u aq[4];
#pragma unroll
  for (int kk = 0; kk < 4; ++kk) aq[kk] = lds_load_a(Qs, LDB, mg * 16, kk * 32, lane);

  float m[8], l[8];
#pragma unroll
  for (int r = 0; r < 8; ++r) { m[r] = -3.0e38f; l[r] = 0.0f; }

  // ---- pass 1: online max / sum-exp -------------------------------------
  {
    const unsigned short* kp = Kb + (size_t)sr * NH + sx;
    const float* bp = bias + (size_t)(t0 + mg * 16 + half8) * T + sg * 16 + (lane & 15);
    for (int s0 = 0; s0 < T; s0 += 64) {
      __syncthreads();
#ifdef HAVE_ASYNC_LDS
#pragma unroll
      for (int i = 0; i < 4; ++i)
        async_cp16(kd + i * 16 * LDB, kp + (size_t)i * 16 * NH);
      async_wait0();
#else
#pragma unroll
      for (int i = 0; i < 16; ++i) {
        int e = t + i * 256, r = e >> 6, cu = e & 63;
        *(unsigned int*)(Ks + r * LDB + cu * 2) =
            *(const unsigned int*)(Kb + (size_t)(s0 + r) * NH + cu * 2);
      }
#endif
      __syncthreads();
      v8f c = {};
#pragma unroll
      for (int kk = 0; kk < 4; ++kk)
        c = wmma_bf16(aq[kk], lds_load_b_nk(Ks, LDB, sg * 16, kk * 32, lane), c);
#pragma unroll
      for (int r = 0; r < 8; ++r) {
        float x = c[r] + bp[(size_t)r * T];
        float mn = fmaxf(m[r], x);
        l[r] = l[r] * __expf(m[r] - mn) + __expf(x - mn);
        m[r] = mn;
      }
      kp += (size_t)64 * NH;
      bp += 64;
    }
  }

  // reduce across the 16 lanes that share a row
#pragma unroll
  for (int r = 0; r < 8; ++r) {
#pragma unroll
    for (int off = 1; off < 16; off <<= 1) {
      float mo = __shfl_xor(m[r], off, 32);
      float lo = __shfl_xor(l[r], off, 32);
      float mn = fmaxf(m[r], mo);
      l[r] = l[r] * __expf(m[r] - mn) + lo * __expf(mo - mn);
      m[r] = mn;
    }
  }
  if ((lane & 15) == 0) {
#pragma unroll
    for (int r = 0; r < 8; ++r) {
      int ri = mg * 16 + r + half8;
      m_red[sg][ri] = m[r];
      l_red[sg][ri] = l[r];
    }
  }
  __syncthreads();
  float mf[8], inv[8];
#pragma unroll
  for (int r = 0; r < 8; ++r) {
    int ri = mg * 16 + r + half8;
    float mm = m_red[0][ri], ll = l_red[0][ri];
#pragma unroll
    for (int s = 1; s < 4; ++s) {
      float mo = m_red[s][ri], lo = l_red[s][ri];
      float mn = fmaxf(mm, mo);
      ll = ll * __expf(mm - mn) + lo * __expf(mo - mn);
      mm = mn;
    }
    mf[r] = mm;
    inv[r] = 1.0f / ll;
  }

  // ---- pass 2: recompute logits, write normalized probs ------------------
  {
    float* Pb = probs + (size_t)(b * 16 + n) * T * T;
    const unsigned short* kp = Kb + (size_t)sr * NH + sx;
    const float* bp = bias + (size_t)(t0 + mg * 16 + half8) * T + sg * 16 + (lane & 15);
    float* pp = Pb + (size_t)(t0 + mg * 16 + half8) * T + sg * 16 + (lane & 15);
    for (int s0 = 0; s0 < T; s0 += 64) {
      __syncthreads();
#ifdef HAVE_ASYNC_LDS
#pragma unroll
      for (int i = 0; i < 4; ++i)
        async_cp16(kd + i * 16 * LDB, kp + (size_t)i * 16 * NH);
      async_wait0();
#else
#pragma unroll
      for (int i = 0; i < 16; ++i) {
        int e = t + i * 256, r = e >> 6, cu = e & 63;
        *(unsigned int*)(Ks + r * LDB + cu * 2) =
            *(const unsigned int*)(Kb + (size_t)(s0 + r) * NH + cu * 2);
      }
#endif
      __syncthreads();
      v8f c = {};
#pragma unroll
      for (int kk = 0; kk < 4; ++kk)
        c = wmma_bf16(aq[kk], lds_load_b_nk(Ks, LDB, sg * 16, kk * 32, lane), c);
#pragma unroll
      for (int r = 0; r < 8; ++r) {
        float x = c[r] + bp[(size_t)r * T];
        pp[(size_t)r * T] = __expf(x - mf[r]) * inv[r];
      }
      kp += (size_t)64 * NH;
      bp += 64;
      pp += 64;
    }
  }
}

// ---------------------------------------------------------------------------
// Kernel 3: context = probs (f32, B,N,T,S) x V (bf16, B,S,NH) -> ctx bf16.
// Tile 128(T) x 128(H), K-step 32 over S.  V staged transposed (H-major).
// ---------------------------------------------------------------------------
__global__ __launch_bounds__(256) void mha_context(
    const float* __restrict__ probs, const unsigned short* __restrict__ V,
    unsigned short* __restrict__ Ctx) {
  constexpr int T = 2048, NH = 2048, H = 128;
  __shared__ __align__(16) unsigned short Ps[128 * LDA];   // T x S
  __shared__ __align__(16) unsigned short Vs[128 * LDA];   // H x S (transposed)
  const int t = threadIdx.x, lane = t & 31, w = t >> 5;
  const int wm = w & 3, wn = w >> 2;
  const int head = blockIdx.y, b = head >> 4, n = head & 15;
  const int m0 = blockIdx.x * 128;
  const float* Pb = probs + (size_t)head * T * T;
  const unsigned short* Vb = V + (size_t)b * T * NH + n * H;
  v8f acc[2][4] = {};

  for (int s0 = 0; s0 < T; s0 += 32) {
    __syncthreads();
#pragma unroll
    for (int i = 0; i < 8; ++i) {  // stage P: 128 x 32, float2 + perm pack
      int e = t + i * 256, r = e >> 4, cp = e & 15;
      float2 v = *(const float2*)(Pb + (size_t)(m0 + r) * T + s0 + cp * 2);
      *(unsigned int*)(Ps + r * LDA + cp * 2) = pk_bf16(v.x, v.y);
    }
#pragma unroll
    for (int i = 0; i < 8; ++i) {  // stage V transposed: Vs[h][s]
      int e = t + i * 256, r = e >> 6, hp = e & 63;
      unsigned int vv = *(const unsigned int*)(Vb + (size_t)(s0 + r) * NH + hp * 2);
      Vs[(2 * hp) * LDA + r] = (unsigned short)(vv & 0xFFFFu);
      Vs[(2 * hp + 1) * LDA + r] = (unsigned short)(vv >> 16);
    }
    __syncthreads();
    v8u a[2], bfr[4];
#pragma unroll
    for (int mi = 0; mi < 2; ++mi) a[mi] = lds_load_a(Ps, LDA, wm * 32 + mi * 16, 0, lane);
#pragma unroll
    for (int ni = 0; ni < 4; ++ni) bfr[ni] = lds_load_b_nk(Vs, LDA, wn * 64 + ni * 16, 0, lane);
#pragma unroll
    for (int mi = 0; mi < 2; ++mi)
#pragma unroll
      for (int ni = 0; ni < 4; ++ni) acc[mi][ni] = wmma_bf16(a[mi], bfr[ni], acc[mi][ni]);
  }

  const int half8 = (lane >> 4) ? 8 : 0;
#pragma unroll
  for (int ni = 0; ni < 4; ++ni) {
    int col = wn * 64 + ni * 16 + (lane & 15);
    unsigned short* cp =
        Ctx + ((size_t)b * T + m0 + wm * 32 + half8) * NH + n * H + col;
#pragma unroll
    for (int mi = 0; mi < 2; ++mi)
#pragma unroll
      for (int r = 0; r < 8; ++r)
        cp[(size_t)(mi * 16 + r) * NH] = f2bf(acc[mi][ni][r]);
  }
}

// ---------------------------------------------------------------------------
// Kernel 4: output projection.  data = ctx (4096 x 2048 bf16) * Wo^T + o_bias.
// Wo is (2048 x 2048) f32, N-major for B fragments.
// ---------------------------------------------------------------------------
__global__ __launch_bounds__(256) void mha_oproj(
    const unsigned short* __restrict__ Ctx, const float* __restrict__ Wo,
    const float* __restrict__ obias, float* __restrict__ Outp) {
  constexpr int K = 2048, N = 2048;
  __shared__ __align__(16) unsigned short As[128 * LDA];
  __shared__ __align__(16) unsigned short Bs[128 * LDA];
  const int t = threadIdx.x, lane = t & 31, w = t >> 5;
  const int wm = w & 3, wn = w >> 2;
  const int m_base = blockIdx.y * 128, n_base = blockIdx.x * 128;
  const int sr = t >> 2, sx = (t & 3) * 8;  // ctx staging: row / 16B column
  v8f acc[2][4] = {};

#ifdef HAVE_ASYNC_LDS
  const unsigned short* ap = Ctx + (size_t)(m_base + sr) * K + sx;
  unsigned short* ad = As + sr * LDA + sx;
#endif

  for (int k0 = 0; k0 < K; k0 += 32) {
    __syncthreads();
#ifdef HAVE_ASYNC_LDS
#pragma unroll
    for (int i = 0; i < 2; ++i)  // stage ctx 128 x 32 bf16, 16B async copies
      async_cp16(ad + i * 64 * LDA, ap + (size_t)i * 64 * K);
#else
#pragma unroll
    for (int i = 0; i < 8; ++i) {
      int e = t + i * 256, r = e >> 4, cu = e & 15;
      *(unsigned int*)(As + r * LDA + cu * 2) =
          *(const unsigned int*)(Ctx + (size_t)(m_base + r) * K + k0 + cu * 2);
    }
#endif
#pragma unroll
    for (int i = 0; i < 8; ++i) {  // stage Wo 128(N) x 32(K), float2 + perm pack
      int e = t + i * 256, r = e >> 4, cp = e & 15;
      float2 v = *(const float2*)(Wo + (size_t)(n_base + r) * K + k0 + cp * 2);
      *(unsigned int*)(Bs + r * LDA + cp * 2) = pk_bf16(v.x, v.y);
    }
    if (k0 + 32 < K)
      __builtin_prefetch(&Wo[(size_t)(n_base + (t >> 1)) * K + k0 + 32 + (t & 1) * 16], 0, 1);
#ifdef HAVE_ASYNC_LDS
    async_wait0();
    ap += 32;
#endif
    __syncthreads();
    v8u a[2], b[4];
#pragma unroll
    for (int mi = 0; mi < 2; ++mi) a[mi] = lds_load_a(As, LDA, wm * 32 + mi * 16, 0, lane);
#pragma unroll
    for (int ni = 0; ni < 4; ++ni) b[ni] = lds_load_b_nk(Bs, LDA, wn * 64 + ni * 16, 0, lane);
#pragma unroll
    for (int mi = 0; mi < 2; ++mi)
#pragma unroll
      for (int ni = 0; ni < 4; ++ni) acc[mi][ni] = wmma_bf16(a[mi], b[ni], acc[mi][ni]);
  }

  const int half8 = (lane >> 4) ? 8 : 0;
#pragma unroll
  for (int ni = 0; ni < 4; ++ni) {
    int col = n_base + wn * 64 + ni * 16 + (lane & 15);
    float bv = obias[col];
    float* op = Outp + (size_t)(m_base + wm * 32 + half8) * N + col;
#pragma unroll
    for (int mi = 0; mi < 2; ++mi)
#pragma unroll
      for (int r = 0; r < 8; ++r)
        op[(size_t)(mi * 16 + r) * N] = acc[mi][ni][r] + bv;
  }
}

// ---------------------------------------------------------------------------
extern "C" void kernel_launch(void* const* d_in, const int* in_sizes, int n_in,
                              void* d_out, int out_size, void* d_ws, size_t ws_size,
                              hipStream_t stream) {
  constexpr int B = 2, T = 2048, D = 2048, NHEAD = 16, H = 128;
  constexpr int M = B * T;        // 4096
  constexpr int NH = NHEAD * H;   // 2048

  const float* query = (const float*)d_in[0];
  const float* abias = (const float*)d_in[1];
  const float* qw = (const float*)d_in[2];
  const float* qb = (const float*)d_in[3];
  const float* kw = (const float*)d_in[4];
  const float* kb = (const float*)d_in[5];
  const float* vw = (const float*)d_in[6];
  const float* vb = (const float*)d_in[7];
  const float* ow = (const float*)d_in[8];
  const float* ob = (const float*)d_in[9];

  float* out = (float*)d_out;                       // data: (B,T,D)
  float* probs = out + (size_t)B * T * D;           // probs: (B,N,T,S)

  unsigned short* ws = (unsigned short*)d_ws;       // bf16 scratch
  unsigned short* Qm = ws;                          // (M, NH)
  unsigned short* Km = ws + (size_t)M * NH;         // (M, NH)
  unsigned short* Vm = ws + (size_t)2 * M * NH;     // (M, NH)
  unsigned short* Cm = ws + (size_t)3 * M * NH;     // (M, NH)

  const float qscale = 0.08838834764831845f;  // 1/sqrt(128)
  dim3 blk(256);

  mha_qkv_proj<<<dim3(NH / 128, M / 128), blk, 0, stream>>>(query, qw, qb, Qm, qscale);
  mha_qkv_proj<<<dim3(NH / 128, M / 128), blk, 0, stream>>>(query, kw, kb, Km, 1.0f);
  mha_qkv_proj<<<dim3(NH / 128, M / 128), blk, 0, stream>>>(query, vw, vb, Vm, 1.0f);

  mha_attn_softmax<<<dim3(T / 32, NHEAD, B), blk, 0, stream>>>(Qm, Km, abias, probs);

  mha_context<<<dim3(T / 128, B * NHEAD), blk, 0, stream>>>(probs, Vm, Cm);

  mha_oproj<<<dim3(D / 128, M / 128), blk, 0, stream>>>(Cm, ow, ob, out);
}
